// TDNN_36146444763929
// MI455X (gfx1250) — matrix operand order repaired and checked by reading, not compiled, not run on hardware
//
#include <hip/hip_runtime.h>

typedef __attribute__((ext_vector_type(16))) __bf16 v16bf;
typedef __attribute__((ext_vector_type(8)))  float  v8f;
typedef __attribute__((ext_vector_type(4)))  unsigned int v4u;
typedef __attribute__((ext_vector_type(8)))  int v8i;
typedef __attribute__((ext_vector_type(4)))  int v4i;

#define BATCH   64
#define SEQ     1000
#define CIN     512
#define COUT    512
#define KWID    5
#define TVALID  996              // SEQ - KWID + 1
#define KDIM    (CIN * KWID)     // 2560 (GEMM K), kappa = kw*512 + c
#define KSTEPS  (KDIM / 32)      // 80 WMMA K-steps

#define BM 128                   // block tile M (time)
#define BN 256                   // block tile N (c_out)
#define LDSA_STRIDE 80           // bytes per A row in LDS: 64 data + 16 pad
#define ATILE_BYTES (BM * LDSA_STRIDE)

static __device__ __forceinline__ unsigned pack_bf16(float lo, float hi) {
    unsigned ul = __float_as_uint(lo);
    unsigned uh = __float_as_uint(hi);
    ul = (ul + 0x7FFFu + ((ul >> 16) & 1u)) >> 16;   // round-to-nearest-even
    uh = (uh + 0x7FFFu + ((uh >> 16) & 1u)) & 0xFFFF0000u;
    return (ul & 0xFFFFu) | uh;
}

// ---- prepass 1: kernel[o][c][kw] (f32) -> wt[o][kw*512+c] (bf16, K-contiguous rows)
__global__ __launch_bounds__(256) void tdnn_wprep(const float* __restrict__ w,
                                                  unsigned short* __restrict__ wt) {
    int idx = blockIdx.x * 256 + threadIdx.x;    // 0 .. COUT*KDIM-1
    int o   = idx / KDIM;
    int kap = idx % KDIM;
    int kw  = kap >> 9;        // / 512
    int c   = kap & 511;
    float f = w[(o * CIN + c) * KWID + kw];
    unsigned u = __float_as_uint(f);
    u = (u + 0x7FFFu + ((u >> 16) & 1u)) >> 16;
    wt[idx] = (unsigned short)u;
}

// ---- prepass 2: x (f32) -> xb (bf16), same layout [B][T][C]
__global__ __launch_bounds__(256) void tdnn_xprep(const float* __restrict__ x,
                                                  unsigned short* __restrict__ xb) {
    size_t idx = ((size_t)blockIdx.x * 256 + threadIdx.x) * 8;  // 8 elements / thread
    const float4* s = (const float4*)(x + idx);
    float4 a = s[0], c = s[1];
    uint4 q;
    q.x = pack_bf16(a.x, a.y); q.y = pack_bf16(a.z, a.w);
    q.z = pack_bf16(c.x, c.y); q.w = pack_bf16(c.z, c.w);
    *(uint4*)(xb + idx) = q;
}

union Frag { uint4 q[2]; v16bf v; };
union AccS { v8f v; float f[8]; };

// shared epilogue: bias + ReLU + store (uniform fast path when tile fully valid)
static __device__ __forceinline__ void epilogue(const v8f (&acc)[4][4],
                                                const float* __restrict__ bias,
                                                float* __restrict__ out,
                                                int b, int t0, int o0,
                                                int mw, int nw, int l16, int lhi) {
    const bool full = (t0 + BM) <= TVALID;
#pragma unroll
    for (int j = 0; j < 4; ++j) {
        const int o  = o0 + nw + j * 16 + l16;
        const float bv = bias[o];
#pragma unroll
        for (int i = 0; i < 4; ++i) {
            const int mbase = mw + i * 16 + lhi * 8;   // C layout: lanes16-31 hold rows M+8
            AccS a; a.v = acc[i][j];
            float* p = out + ((size_t)b * TVALID + (t0 + mbase)) * COUT + o;
            if (full) {
#pragma unroll
                for (int v = 0; v < 8; ++v) {
                    float r = a.f[v] + bv;
                    p[(size_t)v * COUT] = r > 0.f ? r : 0.f;
                }
            } else {
#pragma unroll
                for (int v = 0; v < 8; ++v) {
                    if (t0 + mbase + v < TVALID) {
                        float r = a.f[v] + bv;
                        p[(size_t)v * COUT] = r > 0.f ? r : 0.f;
                    }
                }
            }
        }
    }
}

// =====================================================================
// Path A: bf16 x in scratch, A-tiles DMA'd by the Tensor Data Mover with
// hardware LDS padding, double-buffered, synced via TENSORcnt + barriers.
// =====================================================================
__global__ __launch_bounds__(256) void tdnn_wmma_tdm(const unsigned short* __restrict__ xb,
                                                     const unsigned short* __restrict__ wt,
                                                     const float* __restrict__ bias,
                                                     float* __restrict__ out) {
    __shared__ __align__(16) unsigned char ldsA[2 * ATILE_BYTES];  // 20 KB, double buffer

    const int b    = blockIdx.z;
    const int t0   = blockIdx.y * BM;
    const int o0   = blockIdx.x * BN;

    const int tid  = threadIdx.x;
    const int wave = tid >> 5;
    const int lane = tid & 31;
    const int l16  = lane & 15;
    const int lhi  = lane >> 4;

    const int mw = (wave & 1) * 64;
    const int nw = (wave >> 1) * 64;

    const v8f zero = {};
    v8f acc[4][4];
#pragma unroll
    for (int i = 0; i < 4; ++i)
#pragma unroll
        for (int j = 0; j < 4; ++j) acc[i][j] = zero;

    const unsigned ldsBase = (unsigned)(size_t)(&ldsA[0]);

    // TDM descriptor issue for K-step kk: 2D tile 32 elems x 128 rows of bf16,
    // row stride 512 elems in memory, LDS rows padded 64B -> 80B by TDM
    // (pad_interval=3 -> every 16 DWORDs, pad_amount=3 -> 4 DWORDs).
    // tensor_dim1 = rows remaining in x -> hardware zero-fills OOB rows.
    auto issue_tdm = [&](int kkN) {
        const int kw = kkN >> 4;
        const int c0 = (kkN & 15) << 5;
        const size_t gaddr = (size_t)(xb + ((size_t)b * SEQ + t0 + kw) * CIN + c0);
        const unsigned dim1 = (unsigned)(SEQ - (t0 + kw));
        const unsigned lds  = ldsBase + (unsigned)(kkN & 1) * ATILE_BYTES;
        v4u g0 = { 1u,                                 // count=1, user descriptor
                   lds,                                // lds_addr
                   (unsigned)gaddr,                    // global_addr[31:0]
                   (unsigned)((gaddr >> 32) & 0x01FFFFFFu) | 0x80000000u }; // [56:32] | type=2
        v8i g1 = { (int)((1u << 16) | (1u << 20) | (3u << 22) | (3u << 25)),
                   //     data_size=2B  pad_en     pad_int=16dw  pad_amt=4dw
                   (int)(512u << 16),                  // tensor_dim0 = 512
                   (int)(dim1 << 16),                  // tensor_dim0 hi=0 | tensor_dim1 lo
                   (int)(32u << 16),                   // tensor_dim1 hi=0 | tile_dim0 = 32
                   128,                                // tile_dim1 = 128, tile_dim2 = 0
                   512,                                // tensor_dim0_stride = 512
                   0, 0 };
        v4i g2 = { 0, 0, 0, 0 };
        v4i g3 = { 0, 0, 0, 0 };
        v8i g4 = { 0, 0, 0, 0, 0, 0, 0, 0 };
        __builtin_amdgcn_tensor_load_to_lds(g0, g1, g2, g3, g4, 0);
    };

    if (wave == 0) issue_tdm(0);

    const int aoff0 = lhi * 16;        // A frag LDS offsets (16-bit A layout)
    const int aoff1 = aoff0 + 32;
    const int boff0 = lhi * 16;        // B frag global offsets (16-bit B layout)
    const int boff1 = boff0 + 32;

    for (int kk = 0; kk < KSTEPS; ++kk) {
        if (wave == 0) {
            if (kk + 1 < KSTEPS) {
                issue_tdm(kk + 1);                        // prefetch next tile
                __builtin_amdgcn_s_wait_tensorcnt(1);     // current tile landed
            } else {
                __builtin_amdgcn_s_wait_tensorcnt(0);
            }
        }
        __syncthreads();   // release all waves to read buffer kk&1

        const unsigned char* bufp = ldsA + (kk & 1) * ATILE_BYTES;

        Frag afrag[4];
#pragma unroll
        for (int i = 0; i < 4; ++i) {
            const unsigned char* ab = bufp + (mw + i * 16 + l16) * LDSA_STRIDE;
            afrag[i].q[0] = *(const uint4*)(ab + aoff0);
            afrag[i].q[1] = *(const uint4*)(ab + aoff1);
        }
        Frag bfrag[4];
#pragma unroll
        for (int j = 0; j < 4; ++j) {
            const unsigned char* bb =
                (const unsigned char*)(wt + (size_t)(o0 + nw + j * 16 + l16) * KDIM + kk * 32);
            bfrag[j].q[0] = *(const uint4*)(bb + boff0);
            bfrag[j].q[1] = *(const uint4*)(bb + boff1);
        }

#pragma unroll
        for (int i = 0; i < 4; ++i)
#pragma unroll
            for (int j = 0; j < 4; ++j)
                acc[i][j] = __builtin_amdgcn_wmma_f32_16x16x32_bf16(
                    false, afrag[i].v, false, bfrag[j].v,
                    (short)0, acc[i][j], false, false);

        __syncthreads();   // reads done before this buffer is DMA-overwritten (kk+2)
    }

    epilogue(acc, bias, out, b, t0, o0, mw, nw, l16, lhi);
}

// =====================================================================
// Path B (fallback, small scratch): inline f32->bf16 conversion into LDS.
// =====================================================================
__global__ __launch_bounds__(256) void tdnn_wmma(const float* __restrict__ x,
                                                 const unsigned short* __restrict__ wt,
                                                 const float* __restrict__ bias,
                                                 float* __restrict__ out) {
    __shared__ __align__(16) unsigned char ldsA[ATILE_BYTES];

    const int b    = blockIdx.z;
    const int t0   = blockIdx.y * BM;
    const int o0   = blockIdx.x * BN;

    const int tid  = threadIdx.x;
    const int wave = tid >> 5;
    const int lane = tid & 31;
    const int l16  = lane & 15;
    const int lhi  = lane >> 4;

    const int mw = (wave & 1) * 64;
    const int nw = (wave >> 1) * 64;

    const v8f zero = {};
    v8f acc[4][4];
#pragma unroll
    for (int i = 0; i < 4; ++i)
#pragma unroll
        for (int j = 0; j < 4; ++j) acc[i][j] = zero;

    const int arow  = tid >> 1;
    const int ahalf = tid & 1;

    const int aoff0 = lhi * 16;
    const int aoff1 = aoff0 + 32;
    const int boff0 = lhi * 16;
    const int boff1 = boff0 + 32;

    for (int kk = 0; kk < KSTEPS; ++kk) {
        const int kw = kk >> 4;
        const int c0 = (kk & 15) << 5;
        {
            int trow = t0 + arow + kw;
            trow = trow < (SEQ - 1) ? trow : (SEQ - 1);
            const float4* src =
                (const float4*)(x + ((size_t)b * SEQ + trow) * CIN + c0 + ahalf * 16);
            float4 f0 = src[0], f1 = src[1], f2 = src[2], f3 = src[3];
            uint4 q0, q1;
            q0.x = pack_bf16(f0.x, f0.y); q0.y = pack_bf16(f0.z, f0.w);
            q0.z = pack_bf16(f1.x, f1.y); q0.w = pack_bf16(f1.z, f1.w);
            q1.x = pack_bf16(f2.x, f2.y); q1.y = pack_bf16(f2.z, f2.w);
            q1.z = pack_bf16(f3.x, f3.y); q1.w = pack_bf16(f3.z, f3.w);
            uint4* dst = (uint4*)(ldsA + arow * LDSA_STRIDE + ahalf * 32);
            dst[0] = q0;
            dst[1] = q1;
        }
        __syncthreads();

        Frag afrag[4];
#pragma unroll
        for (int i = 0; i < 4; ++i) {
            const unsigned char* ab = ldsA + (mw + i * 16 + l16) * LDSA_STRIDE;
            afrag[i].q[0] = *(const uint4*)(ab + aoff0);
            afrag[i].q[1] = *(const uint4*)(ab + aoff1);
        }
        Frag bfrag[4];
#pragma unroll
        for (int j = 0; j < 4; ++j) {
            const unsigned char* bb =
                (const unsigned char*)(wt + (size_t)(o0 + nw + j * 16 + l16) * KDIM + kk * 32);
            bfrag[j].q[0] = *(const uint4*)(bb + boff0);
            bfrag[j].q[1] = *(const uint4*)(bb + boff1);
        }

#pragma unroll
        for (int i = 0; i < 4; ++i)
#pragma unroll
            for (int j = 0; j < 4; ++j)
                acc[i][j] = __builtin_amdgcn_wmma_f32_16x16x32_bf16(
                    false, afrag[i].v, false, bfrag[j].v,
                    (short)0, acc[i][j], false, false);

        __syncthreads();
    }

    epilogue(acc, bias, out, b, t0, o0, mw, nw, l16, lhi);
}

extern "C" void kernel_launch(void* const* d_in, const int* in_sizes, int n_in,
                              void* d_out, int out_size, void* d_ws, size_t ws_size,
                              hipStream_t stream) {
    (void)in_sizes; (void)n_in; (void)out_size;
    const float* x    = (const float*)d_in[0];
    const float* w    = (const float*)d_in[1];
    const float* bias = (const float*)d_in[2];
    float* out        = (float*)d_out;

    const size_t WT_BYTES = (size_t)COUT * KDIM * 2;          // 2.62 MB
    const size_t XB_BYTES = (size_t)BATCH * SEQ * CIN * 2;    // 65.5 MB

    unsigned short* wt = (unsigned short*)d_ws;
    tdnn_wprep<<<dim3((COUT * KDIM) / 256), dim3(256), 0, stream>>>(w, wt);

    dim3 grid(COUT / BN, (TVALID + BM - 1) / BM, BATCH);
    if (ws_size >= WT_BYTES + XB_BYTES) {
        unsigned short* xb = (unsigned short*)((char*)d_ws + WT_BYTES);
        tdnn_xprep<<<dim3((BATCH * SEQ * CIN) / (256 * 8)), dim3(256), 0, stream>>>(x, xb);
        tdnn_wmma_tdm<<<grid, dim3(256), 0, stream>>>(xb, wt, bias, out);
    } else {
        tdnn_wmma<<<grid, dim3(256), 0, stream>>>(x, wt, bias, out);
    }
}
